// LFFT_37658273251872
// MI455X (gfx1250) — matrix operand order repaired and checked by reading, not compiled, not run on hardware
//
#include <hip/hip_runtime.h>
#include <hip/hip_bf16.h>

// ---------------------------------------------------------------------------
// Problem constants (from reference)
// ---------------------------------------------------------------------------
#define T_LEN    2048
#define D_DIM    512
#define V_DIM    32000
#define N_LAYERS 4
#define KCHUNKS  16          // D / 32
#define N_TILES  2000        // V / 16
#define M_TILES  128         // T / 16

typedef __attribute__((ext_vector_type(16))) __bf16 v16bf;
typedef __attribute__((ext_vector_type(8)))  float  v8f;
typedef __attribute__((ext_vector_type(4)))  int    v4i;

#if __has_builtin(__builtin_amdgcn_global_load_async_to_lds_b128) && \
    __has_builtin(__builtin_amdgcn_s_wait_asynccnt)
#define LFFT_USE_ASYNC 1
#else
#define LFFT_USE_ASYNC 0
#endif

#if LFFT_USE_ASYNC
typedef __attribute__((address_space(1))) v4i* lfft_gptr;
typedef __attribute__((address_space(3))) v4i* lfft_lptr;
// global -> LDS 16B async copy, no VGPR data path (ASYNCcnt tracked)
__device__ __forceinline__ void lfft_async_cp16(const void* g, void* l) {
    __builtin_amdgcn_global_load_async_to_lds_b128(
        (lfft_gptr)(unsigned long long)g,
        (lfft_lptr)(unsigned int)(unsigned long long)l,
        0, 0);
}
#endif

// float -> bf16 (round-to-nearest-even), also return the residual
__device__ inline unsigned short bf16_rne(float f, float* rem) {
    unsigned int u = __float_as_uint(f);
    unsigned int v = u + (0x7FFFu + ((u >> 16) & 1u));
    unsigned short h = (unsigned short)(v >> 16);
    *rem = f - __uint_as_float(((unsigned int)h) << 16);
    return h;
}

// ---------------------------------------------------------------------------
// Kernel 1: Wsum[l][d] = sum_k hash_W[l][k][d]   (codes.sum(-1) == h . Wsum)
// ---------------------------------------------------------------------------
__global__ void lfft_wsum_kernel(const float* __restrict__ hashW,
                                 float* __restrict__ Wsum) {
    int i = blockIdx.x * blockDim.x + threadIdx.x;   // 0 .. 4*512-1
    if (i >= N_LAYERS * D_DIM) return;
    int l = i / D_DIM, d = i % D_DIM;
    float s = 0.f;
    #pragma unroll
    for (int k = 0; k < 16; ++k)
        s += hashW[((size_t)(l * 16 + k)) * D_DIM + d];
    Wsum[i] = s;
}

// ---------------------------------------------------------------------------
// Kernel 2: pack out_W into WMMA B-fragments, bf16 hi/lo INTERLEAVED:
//   Bpk[nTile][c][hl][lane][16],  hl = 0 (bf16 hi) / 1 (bf16 residual).
// B (32x16, KxN) layout used here (contiguous K, same as A): lane 0-15 -> N=lane, K=e (e=0..15);
//                                lane 16-31 -> N=lane-16, K=16+e.
// ---------------------------------------------------------------------------
__global__ void lfft_packB_kernel(const float* __restrict__ W,
                                  unsigned short* __restrict__ Bpk) {
    int tid = blockIdx.x * blockDim.x + threadIdx.x;  // 0 .. 2000*16*32-1
    if (tid >= N_TILES * KCHUNKS * 32) return;
    int lane  = tid & 31;
    int c     = (tid >> 5) & (KCHUNKS - 1);
    int nTile = tid >> 9;
    int col   = nTile * 16 + (lane & 15);
    int khalf = (lane >> 4) * 16;

    unsigned short hb[16], lb[16];
    #pragma unroll
    for (int e = 0; e < 16; ++e) {
        int d = c * 32 + khalf + e;
        float f = W[(size_t)d * V_DIM + col];
        float rem, r2;
        hb[e] = bf16_rne(f, &rem);
        lb[e] = bf16_rne(rem, &r2);
    }
    size_t baseHi = (((size_t)(nTile * KCHUNKS + c) * 2 + 0) * 32 + lane) * 16;
    unsigned int* ph = (unsigned int*)(Bpk + baseHi);
    unsigned int* pl = (unsigned int*)(Bpk + baseHi + 512);   // hl stride = 32*16
    #pragma unroll
    for (int q = 0; q < 8; ++q) {
        ph[q] = (unsigned int)hb[2*q] | ((unsigned int)hb[2*q+1] << 16);
        pl[q] = (unsigned int)lb[2*q] | ((unsigned int)lb[2*q+1] << 16);
    }
}

// ---------------------------------------------------------------------------
// Kernel 3: features + 4 liquid/wave layers for one t (h identical over batch),
// then write h as bf16 hi/lo interleaved in WMMA A-fragment layout:
//   Apk[mTile][c][hl][lane][16]
// A (16x32, MxK): lane 0-15 -> M=lane, K=e (e=0..15);
//                lanes 16-31 -> M=lane-16, K=16+e.
// ---------------------------------------------------------------------------
__global__ void lfft_feat_layers_kernel(const float* __restrict__ Wsum,
                                        const float* __restrict__ efreqs,
                                        const float* __restrict__ eamps,
                                        const float* __restrict__ wfreqs,
                                        const float* __restrict__ wphases,
                                        const float* __restrict__ wamps,
                                        unsigned short* __restrict__ Apk) {
    const int t   = blockIdx.x;      // 0..2047
    const int tid = threadIdx.x;     // 0..127
    __shared__ float waves[16];
    __shared__ float red[128];
    __shared__ float modsh;

    const float TWO_PI = 6.28318530717958647692f;
    const float tf = (float)t;
    float hreg[4];

    // ---- frequency features: d = i*128 + (cos?64:0) + f ; freq = 10^(0.5i-1+2f/63)
    #pragma unroll
    for (int j = 0; j < 4; ++j) {
        int d = tid * 4 + j;
        int i = d >> 7;
        int r = d & 127;
        int f = r & 63;
        float expo = 0.5f * (float)i - 1.0f + (2.0f / 63.0f) * (float)f;
        float freq = exp10f(expo);
        float arg  = tf * freq * (TWO_PI / (float)T_LEN);
        hreg[j] = (r >= 64) ? cosf(arg) : sinf(arg);
    }
    const float tnorm = tf * (TWO_PI / (float)T_LEN);

    for (int l = 0; l < N_LAYERS; ++l) {
        if (tid < 16)
            waves[tid] = sinf(tf * wfreqs[l * 16 + tid] + wphases[l * 16 + tid]);
        // h . Wsum[l]
        float p = 0.f;
        #pragma unroll
        for (int j = 0; j < 4; ++j)
            p += hreg[j] * Wsum[l * D_DIM + tid * 4 + j];
        red[tid] = p;
        __syncthreads();
        for (int s = 64; s > 0; s >>= 1) {
            if (tid < s) red[tid] += red[tid + s];
            __syncthreads();
        }
        if (tid == 0) {
            float m = fmodf(fabsf(red[0]), 32.0f);
            int idx = (int)m;
            if (idx > 31) idx = 31;
            float mod = 0.f;
            #pragma unroll
            for (int k = 0; k < 8; ++k) {
                int o = (l * 32 + idx) * 8 + k;
                mod += eamps[o] * sinf(efreqs[o] * tnorm);
            }
            modsh = mod;
        }
        __syncthreads();
        // h = h*(1.5 + 0.05*mod) + 0.5 * (waves @ wa)
        float scale = 1.5f + 0.05f * modsh;
        #pragma unroll
        for (int j = 0; j < 4; ++j) {
            int d = tid * 4 + j;
            float I = 0.f;
            #pragma unroll
            for (int w = 0; w < 16; ++w)
                I += waves[w] * wamps[((size_t)(l * 16 + w)) * D_DIM + d];
            hreg[j] = hreg[j] * scale + 0.5f * I;
        }
        __syncthreads();
    }

    // ---- pack into A fragments (hi/lo bf16 interleaved)
    int mTile = t >> 4;
    int lih   = t & 15;
    #pragma unroll
    for (int j = 0; j < 4; ++j) {
        int d  = tid * 4 + j;
        int c  = d >> 5;
        int kk = d & 31;
        int lh = kk >> 4;                             // lane half (K contiguous)
        int e  = kk & 15;                             // element in fragment
        int lane = lih + lh * 16;
        size_t off = (((size_t)(mTile * KCHUNKS + c) * 2 + 0) * 32 + lane) * 16 + e;
        float rem, r2;
        unsigned short hb = bf16_rne(hreg[j], &rem);
        unsigned short lb = bf16_rne(rem, &r2);
        Apk[off]       = hb;       // hl=0
        Apk[off + 512] = lb;       // hl=1 (stride 32 lanes * 16 elems)
    }
}

// ---------------------------------------------------------------------------
// Kernel 4: LDS-tiled WMMA GEMM, bf16x3 (Ahi*Bhi + Ahi*Blo + Alo*Bhi, f32 acc).
// Block = 256 threads = 8 waves in a 2(M) x 4(N) grid.
// Block tile = 64(M) x 128(N); each wave computes 2 mTiles x 2 nTiles.
// Double-buffered LDS (48 KB).  Staging uses GLOBAL_LOAD_ASYNC_TO_LDS_B128
// (no VGPR data path, ASYNCcnt) so nothing is live across the WMMA region.
// Per chunk: wait_async(0) -> barrier -> issue async stage of c+1 into the
// other buffer -> WMMAs consume chunk c from LDS while the copies fly.
// ---------------------------------------------------------------------------
__global__ __launch_bounds__(256, 1) void
lfft_gemm_kernel(const unsigned short* __restrict__ Apk,
                 const unsigned short* __restrict__ Bpk,
                 const float* __restrict__ bias,
                 float* __restrict__ out) {
    const int tid  = threadIdx.x;
    const int lane = tid & 31;
    const int w    = tid >> 5;          // wave 0..7
    const int wm   = w >> 2;            // 0..1  (M direction)
    const int wn   = w & 3;             // 0..3  (N direction)
    const int mTile0 = blockIdx.y * 4;  // block covers 4 mTiles (64 rows)
    const int nTile0 = blockIdx.x * 8;  // block covers 8 nTiles (128 cols)

    // [buf][tile][hl][lane][16] ; A: 4 mTiles, B: 8 nTiles, double buffered
    __shared__ alignas(32) unsigned short sA[2 * 4 * 1024];   // 16 KB
    __shared__ alignas(32) unsigned short sB[2 * 8 * 1024];   // 32 KB

    const uint4* gA = (const uint4*)Apk;   // uniform SGPR base
    const uint4* gB = (const uint4*)Bpk;

    // stage chunk c into LDS buffer `buf` (24 KB total, 6 x 16B per thread)
    auto stage = [&](int c, int buf) {
        #pragma unroll
        for (int r = 0; r < 2; ++r) {                  // 512 uint4 of A
            unsigned flat = (unsigned)tid + 256u * r;
            unsigned i = flat >> 7, q = flat & 127u;   // mTile piece, uint4-in-piece
            const uint4* src = gA + (unsigned)((mTile0 + i) * KCHUNKS + c) * 128u + q;
            unsigned short* dst = sA + buf * 4096 + flat * 8;
#if LFFT_USE_ASYNC
            lfft_async_cp16(src, dst);
#else
            *(uint4*)dst = *src;
#endif
        }
        #pragma unroll
        for (int r = 0; r < 4; ++r) {                  // 1024 uint4 of B
            unsigned flat = (unsigned)tid + 256u * r;
            unsigned j = flat >> 7, q = flat & 127u;   // nTile piece
            const uint4* src = gB + (unsigned)((nTile0 + j) * KCHUNKS + c) * 128u + q;
            unsigned short* dst = sB + buf * 8192 + flat * 8;
#if LFFT_USE_ASYNC
            lfft_async_cp16(src, dst);
#else
            *(uint4*)dst = *src;
#endif
        }
    };

    v8f acc[2][2];
    #pragma unroll
    for (int i = 0; i < 2; ++i)
        #pragma unroll
        for (int j = 0; j < 2; ++j) acc[i][j] = (v8f)0.0f;

    stage(0, 0);

    for (int c = 0; c < KCHUNKS; ++c) {
#if LFFT_USE_ASYNC
        __builtin_amdgcn_s_wait_asynccnt(0);   // my chunk-c copies are in LDS
#endif
        __syncthreads();                       // everyone's chunk-c copies visible;
                                               // all chunk-(c-1) readers done
        if (c + 1 < KCHUNKS)
            stage(c + 1, (c + 1) & 1);         // fills the other buffer: race-free

        const v16bf* fA = (const v16bf*)(sA + (c & 1) * 4096);
        const v16bf* fB = (const v16bf*)(sB + (c & 1) * 8192);
        v16bf ah[2], al[2];
        #pragma unroll
        for (int i = 0; i < 2; ++i) {
            int mt = wm * 2 + i;
            ah[i] = fA[(mt * 2 + 0) * 32 + lane];
            al[i] = fA[(mt * 2 + 1) * 32 + lane];
        }
        #pragma unroll
        for (int j = 0; j < 2; ++j) {
            int ntl = wn * 2 + j;
            v16bf bh = fB[(ntl * 2 + 0) * 32 + lane];
            v16bf bl = fB[(ntl * 2 + 1) * 32 + lane];
            #pragma unroll
            for (int i = 0; i < 2; ++i) {
                acc[i][j] = __builtin_amdgcn_wmma_f32_16x16x32_bf16(
                    false, al[i], false, bh, (short)0, acc[i][j], false, false);
                acc[i][j] = __builtin_amdgcn_wmma_f32_16x16x32_bf16(
                    false, ah[i], false, bl, (short)0, acc[i][j], false, false);
                acc[i][j] = __builtin_amdgcn_wmma_f32_16x16x32_bf16(
                    false, ah[i], false, bh, (short)0, acc[i][j], false, false);
            }
        }
    }

    // C/D layout: VGPR r -> M = r (lanes 0-15) / r+8 (lanes 16-31); N = lane%16
    #pragma unroll
    for (int i = 0; i < 2; ++i) {
        int mTile = mTile0 + wm * 2 + i;
        int trowBase = mTile * 16 + ((lane >> 4) << 3);
        #pragma unroll
        for (int j = 0; j < 2; ++j) {
            int col = (nTile0 + wn * 2 + j) * 16 + (lane & 15);
            float bb = bias[col];
            #pragma unroll
            for (int r = 0; r < 8; ++r) {
                int trow = trowBase + r;
                float v = acc[i][j][r] + bb;
                out[(size_t)trow * V_DIM + col] = v;               // batch 0
                out[((size_t)trow + T_LEN) * V_DIM + col] = v;     // batch 1
            }
        }
    }
}

// ---------------------------------------------------------------------------
extern "C" void kernel_launch(void* const* d_in, const int* in_sizes, int n_in,
                              void* d_out, int out_size, void* d_ws, size_t ws_size,
                              hipStream_t stream) {
    // inputs: 0=x(unused) 1=hash_W 2=expert_freqs 3=expert_amps
    //         4=wave_freqs 5=wave_phases 6=wave_amps 7=out_W 8=out_b
    const float* hashW   = (const float*)d_in[1];
    const float* efreqs  = (const float*)d_in[2];
    const float* eamps   = (const float*)d_in[3];
    const float* wfreqs  = (const float*)d_in[4];
    const float* wphases = (const float*)d_in[5];
    const float* wamps   = (const float*)d_in[6];
    const float* outW    = (const float*)d_in[7];
    const float* outB    = (const float*)d_in[8];
    float* out = (float*)d_out;

    // workspace carving (all offsets 32B aligned)
    char* ws = (char*)d_ws;
    size_t off = 0;
    unsigned short* Bpk = (unsigned short*)(ws + off);
    off += (size_t)D_DIM * V_DIM * 2 * 2;                    // 64 MB (hi+lo)
    unsigned short* Apk = (unsigned short*)(ws + off);
    off += (size_t)T_LEN * D_DIM * 2 * 2;                    //  4 MB (hi+lo)
    float* Wsum = (float*)(ws + off);
    off += (size_t)N_LAYERS * D_DIM * 4;

    // 1) Wsum
    lfft_wsum_kernel<<<(N_LAYERS * D_DIM + 255) / 256, 256, 0, stream>>>(hashW, Wsum);

    // 2) pack out_W -> interleaved bf16 hi/lo B fragments
    {
        int total = N_TILES * KCHUNKS * 32;
        lfft_packB_kernel<<<(total + 255) / 256, 256, 0, stream>>>(outW, Bpk);
    }

    // 3) features + layers -> interleaved bf16 hi/lo A fragments
    lfft_feat_layers_kernel<<<T_LEN, 128, 0, stream>>>(
        Wsum, efreqs, eamps, wfreqs, wphases, wamps, Apk);

    // 4) LDS-tiled WMMA GEMM + bias, duplicated over the two identical batches
    {
        dim3 grid(N_TILES / 8, M_TILES / 4);   // (250, 32) -> 8000 blocks
        lfft_gemm_kernel<<<grid, 256, 0, stream>>>(Apk, Bpk, outB, out);
    }
}
